// Model_58909771432742
// MI455X (gfx1250) — matrix operand order repaired and checked by reading, not compiled, not run on hardware
//
#include <hip/hip_runtime.h>
#include <hip/hip_bf16.h>

typedef __attribute__((ext_vector_type(16))) _Float16 v16h;
typedef __attribute__((ext_vector_type(8)))  float    v8f;

#define IN_FEATS 64
#define HD 128          // NUM_HEADS * ATTN_FEATS
#define NUM_HEADS 4
#define ATTN_FEATS 32
#define N_REL 8
#define NEG_SLOPE 0.2f

// ---------- ordered-float helpers for atomic max on signed floats ----------
__device__ __forceinline__ unsigned f2ord(float f) {
  unsigned b = __float_as_uint(f);
  return (b & 0x80000000u) ? ~b : (b | 0x80000000u);
}
__device__ __forceinline__ float ord2f(unsigned o) {
  unsigned b = (o & 0x80000000u) ? (o ^ 0x80000000u) : ~o;
  return __uint_as_float(b);
}

// ---------- K_W: repack f32 weights into f16 WMMA B-fragment layout ----------
// B is KxN = 32x16 per (rel, ntile, kstep). ISA layout: lane l holds column
// n = l&15; half = l>>4; element j holds W[32*ks + 8*half + (j<8 ? j : j+8)][n].
// Buffer layout: frag[r][nt][lane][ks][16] -> each lane's (ks0,ks1) data is
// 64 contiguous bytes => one clause of 4x global_load_b128.
__global__ void narre_prep_weights(const float* __restrict__ Wsrc,
                                   const float* __restrict__ Wqual,
                                   _Float16* __restrict__ Bsrc,
                                   _Float16* __restrict__ Bqual) {
  int t = blockIdx.x * blockDim.x + threadIdx.x;   // 8*8*2*32 = 4096 threads
  if (t >= N_REL * 8 * 2 * 32) return;
  int lane = t & 31;
  int rest = t >> 5;
  int ks = rest & 1; rest >>= 1;
  int nt = rest & 7;
  int r  = rest >> 3;
  int n    = lane & 15;
  int half = lane >> 4;
  size_t outBase = ((((size_t)(r * 8 + nt) * 32 + lane) * 2) + ks) * 16;
#pragma unroll
  for (int j = 0; j < 16; ++j) {
    int k = 32 * ks + 8 * half + ((j < 8) ? j : (j + 8));
    size_t widx = (size_t)r * IN_FEATS * HD + (size_t)k * HD + nt * 16 + n;
    Bsrc[outBase + j]  = (_Float16)Wsrc[widx];
    Bqual[outBase + j] = (_Float16)Wqual[widx];
  }
}

// ---------- K1: relation histogram ----------
__global__ void narre_hist(const int* __restrict__ rtype, int* __restrict__ cnt, int E) {
  int e = blockIdx.x * blockDim.x + threadIdx.x;
  if (e < E) atomicAdd(&cnt[rtype[e]], 1);
}

// ---------- K2: padded (16-aligned) prefix sum + cursor init ----------
__global__ void narre_prefix(const int* __restrict__ cnt, int* __restrict__ offs,
                             int* __restrict__ cur) {
  if (threadIdx.x == 0 && blockIdx.x == 0) {
    int acc = 0;
    for (int r = 0; r < N_REL; ++r) {
      offs[r] = acc;
      cur[r] = acc;
      acc += ((cnt[r] + 15) >> 4) << 4;      // align each bucket to 16 edges
    }
    offs[N_REL] = acc;
  }
}

// ---------- K3: scatter edges into relation buckets ----------
__global__ void narre_scatter(const int* __restrict__ rtype, int* __restrict__ cur,
                              int* __restrict__ perm, int E) {
  int e = blockIdx.x * blockDim.x + threadIdx.x;
  if (e < E) {
    int pos = atomicAdd(&cur[rtype[e]], 1);
    perm[pos] = e;
  }
}

// ---------- K4: fused projection + leaky-relu + attn dot, via WMMA ----------
// One wave32 handles a tile of 16 edges of one relation.
__global__ void __launch_bounds__(256)
narre_score_wmma(const float* __restrict__ feat, const float* __restrict__ qual,
                 const _Float16* __restrict__ Bsrc, const _Float16* __restrict__ Bqual,
                 const float* __restrict__ bias_s, const float* __restrict__ bias_q,
                 const float* __restrict__ attn,
                 const int* __restrict__ srcIdx, const int* __restrict__ nidIdx,
                 const int* __restrict__ dstIdx,
                 const int* __restrict__ perm, const int* __restrict__ offs,
                 float* __restrict__ score, unsigned* __restrict__ maxord,
                 int numTiles) {
  int wave = blockIdx.x * (blockDim.x >> 5) + (threadIdx.x >> 5);
  if (wave >= numTiles) return;                    // wave-uniform
  int lane = threadIdx.x & 31;
  int base = wave << 4;
  int total = offs[N_REL];
  if (base >= total) return;                       // wave-uniform

  // find relation for this tile (buckets are 16-aligned, no straddle)
  int r = 0;
#pragma unroll
  for (int i = 1; i < N_REL; ++i) r = (base >= offs[i]) ? i : r;

  int half = lane >> 4;
  int m16  = lane & 15;

  // per-lane A rows (edges of this tile); padding slots are perm==-1
  int p = perm[base + m16];
  int rowS = (p >= 0) ? srcIdx[p] : 0;
  int rowQ = (p >= 0) ? nidIdx[p] : 0;

  // A fragments: lane holds row m=lane&15; k = (j<8? j : j+8) + 8*half + 32*ks
  union V16 { v16h v; _Float16 h[16]; };
  V16 aS[2], aQ[2];
#pragma unroll
  for (int ks = 0; ks < 2; ++ks) {
    const float* bs = feat + (size_t)rowS * IN_FEATS + ks * 32 + half * 8;
    const float* bq = qual + (size_t)rowQ * IN_FEATS + ks * 32 + half * 8;
#pragma unroll
    for (int j = 0; j < 16; ++j) {
      int o = (j < 8) ? j : (j + 8);
      aS[ks].h[j] = (_Float16)bs[o];
      aQ[ks].h[j] = (_Float16)bq[o];
    }
  }

  float scoreAcc[8];
#pragma unroll 2
  for (int nt = 0; nt < 8; ++nt) {                 // 8 N-tiles of 16 (= HD 128)
    // Issue all four fragment loads (2x 64B contiguous per matrix) up front so
    // the scheduler can clause them and overlap with neighboring WMMAs.
    const v16h* pS = (const v16h*)Bsrc  + ((size_t)(r * 8 + nt) * 64 + lane * 2);
    const v16h* pQ = (const v16h*)Bqual + ((size_t)(r * 8 + nt) * 64 + lane * 2);
    v16h bS0 = pS[0];
    v16h bS1 = pS[1];
    v16h bQ0 = pQ[0];
    v16h bQ1 = pQ[1];

    union V8 { v8f v; float f[8]; } c;
#pragma unroll
    for (int g = 0; g < 8; ++g) c.f[g] = 0.0f;
    c.v = __builtin_amdgcn_wmma_f32_16x16x32_f16(false, aS[0].v, false, bS0,
                                                 (short)0, c.v, false, false);
    c.v = __builtin_amdgcn_wmma_f32_16x16x32_f16(false, aS[1].v, false, bS1,
                                                 (short)0, c.v, false, false);
    c.v = __builtin_amdgcn_wmma_f32_16x16x32_f16(false, aQ[0].v, false, bQ0,
                                                 (short)0, c.v, false, false);
    c.v = __builtin_amdgcn_wmma_f32_16x16x32_f16(false, aQ[1].v, false, bQ1,
                                                 (short)0, c.v, false, false);

    // C/D layout: VGPR g, lane: row m = g + 8*half, col n = lane&15
    int o = nt * 16 + m16;                          // output feature index
    float bias = bias_s[r * HD + o] + bias_q[r * HD + o];
    float av   = attn[r * HD + o];                  // attn[r][h][d], h*32+d == o

#pragma unroll
    for (int g = 0; g < 8; ++g) {
      float x = c.f[g] + bias;
      x = (x > 0.0f) ? x : NEG_SLOPE * x;
      x *= av;
      // reduce over the 16 columns (within the lane-half)
      x += __shfl_xor(x, 1, 32);
      x += __shfl_xor(x, 2, 32);
      x += __shfl_xor(x, 4, 32);
      x += __shfl_xor(x, 8, 32);
      if (nt & 1) scoreAcc[g] += x; else scoreAcc[g] = x;
    }

    if (nt & 1) {                                   // a full head (2 n-tiles) done
      int h = nt >> 1;
      if (m16 == 0) {                               // lanes 0 and 16 emit
#pragma unroll
        for (int g = 0; g < 8; ++g) {
          int rowm = g + 8 * half;
          int pe = perm[base + rowm];
          if (pe >= 0) {
            float v = scoreAcc[g];
            score[(size_t)pe * NUM_HEADS + h] = v;
            int d = dstIdx[pe];
            atomicMax(&maxord[(size_t)d * NUM_HEADS + h], f2ord(v));
          }
        }
      }
    }
  }
}

// ---------- K5: p = exp(score - max[dst]); denom[dst] += p ----------
__global__ void narre_exp(float* __restrict__ score, const int* __restrict__ dstIdx,
                          const unsigned* __restrict__ maxord,
                          float* __restrict__ denom, int E) {
  int idx = blockIdx.x * blockDim.x + threadIdx.x;
  if (idx >= E * NUM_HEADS) return;
  int e = idx >> 2;
  int h = idx & 3;
  int d = dstIdx[e];
  float m = ord2f(maxord[(size_t)d * NUM_HEADS + h]);
  float pv = __expf(score[idx] - m);
  score[idx] = pv;
  atomicAdd(&denom[(size_t)d * NUM_HEADS + h], pv);
}

// ---------- K6: ft[dst, h, :] += (p/denom) * feat[src, :] ----------
__global__ void __launch_bounds__(256)
narre_message(const float* __restrict__ feat, const int* __restrict__ srcIdx,
              const int* __restrict__ dstIdx, const float* __restrict__ score,
              const float* __restrict__ denom, float* __restrict__ out, int E) {
  int e = blockIdx.x;
  if (e >= E) return;
  int t = threadIdx.x;          // 256 = 4 heads * 64 feats
  int h = t >> 6;
  int i = t & 63;
  int s = srcIdx[e];
  int d = dstIdx[e];
  float a = score[(size_t)e * NUM_HEADS + h] / denom[(size_t)d * NUM_HEADS + h];
  float val = feat[(size_t)s * IN_FEATS + i] * a;
  atomicAdd(&out[(size_t)d * (NUM_HEADS * IN_FEATS) + h * IN_FEATS + i], val);
}

extern "C" void kernel_launch(void* const* d_in, const int* in_sizes, int n_in,
                              void* d_out, int out_size, void* d_ws, size_t ws_size,
                              hipStream_t stream) {
  const float* feat    = (const float*)d_in[0];
  const float* qualemb = (const float*)d_in[1];
  const float* Wsrc    = (const float*)d_in[2];
  const float* bias_s  = (const float*)d_in[3];
  const float* Wqual   = (const float*)d_in[4];
  const float* bias_q  = (const float*)d_in[5];
  const float* attn    = (const float*)d_in[6];
  const int*   src     = (const int*)d_in[7];
  const int*   dst     = (const int*)d_in[8];
  const int*   rtype   = (const int*)d_in[9];
  const int*   nid     = (const int*)d_in[10];
  float* out = (float*)d_out;

  const int N = in_sizes[0] / IN_FEATS;
  const int E = in_sizes[7];

  // ---- workspace layout ----
  char* ws = (char*)d_ws;
  const size_t szB = (size_t)N_REL * 8 * 2 * 32 * 16 * sizeof(_Float16); // 128 KB
  size_t offBsrc  = 0;
  size_t offBqual = offBsrc + szB;
  size_t offCnt   = offBqual + szB;           // 8 ints
  size_t offOffs  = offCnt + 64;              // 9 ints
  size_t offCur   = offOffs + 64;             // 8 ints
  size_t offPerm  = offCnt + 256;
  const int permSlots = ((E + 135) / 16) * 16;   // each bucket padded to 16
  size_t offScore = offPerm + (size_t)permSlots * sizeof(int);
  size_t offMax   = offScore + (size_t)E * NUM_HEADS * sizeof(float);
  size_t offDen   = offMax + (size_t)N * NUM_HEADS * sizeof(unsigned);
  size_t needed   = offDen + (size_t)N * NUM_HEADS * sizeof(float);
  if (ws_size < needed) return;               // deterministic no-op if scratch too small

  _Float16* Bsrc  = (_Float16*)(ws + offBsrc);
  _Float16* Bqual = (_Float16*)(ws + offBqual);
  int* cnt  = (int*)(ws + offCnt);
  int* offs = (int*)(ws + offOffs);
  int* cur  = (int*)(ws + offCur);
  int* perm = (int*)(ws + offPerm);
  float*    score  = (float*)(ws + offScore);
  unsigned* maxord = (unsigned*)(ws + offMax);
  float*    denom  = (float*)(ws + offDen);

  // ---- init (graph-capture-safe async memsets) ----
  hipMemsetAsync(ws + offCnt, 0, 256, stream);                    // cnt/offs/cur
  hipMemsetAsync(perm, 0xFF, (size_t)permSlots * sizeof(int), stream);  // -1
  hipMemsetAsync(maxord, 0, (size_t)N * NUM_HEADS * sizeof(unsigned), stream);
  hipMemsetAsync(denom, 0, (size_t)N * NUM_HEADS * sizeof(float), stream);
  hipMemsetAsync(out, 0, (size_t)out_size * sizeof(float), stream);

  // ---- pipeline ----
  narre_prep_weights<<<(N_REL * 8 * 2 * 32 + 255) / 256, 256, 0, stream>>>(
      Wsrc, Wqual, Bsrc, Bqual);
  narre_hist<<<(E + 255) / 256, 256, 0, stream>>>(rtype, cnt, E);
  narre_prefix<<<1, 32, 0, stream>>>(cnt, offs, cur);
  narre_scatter<<<(E + 255) / 256, 256, 0, stream>>>(rtype, cur, perm, E);

  const int numTiles = permSlots / 16;
  narre_score_wmma<<<(numTiles + 7) / 8, 256, 0, stream>>>(
      feat, qualemb, Bsrc, Bqual, bias_s, bias_q, attn,
      src, nid, dst, perm, offs, score, maxord, numTiles);

  narre_exp<<<(E * NUM_HEADS + 255) / 256, 256, 0, stream>>>(
      score, dst, maxord, denom, E);

  narre_message<<<E, 256, 0, stream>>>(feat, src, dst, score, denom, out, E);
}
